// WeakNTXentLoss_84602265797103
// MI455X (gfx1250) — matrix operand order repaired and checked by reading, not compiled, not run on hardware
//
#include <hip/hip_runtime.h>
#include <hip/hip_bf16.h>

typedef _Float16 f16;
typedef __attribute__((ext_vector_type(16))) _Float16 v16h;
typedef __attribute__((ext_vector_type(8)))  _Float16 v8h;
typedef __attribute__((ext_vector_type(8)))  float    v8f;
typedef __attribute__((ext_vector_type(4)))  float    v4f;
typedef __attribute__((ext_vector_type(4)))  int      v4i;

#define B_ROWS 4096
#define N_ROWS 8192
#define DIM 256
#define KSTEPS 8             // 256 / 32
#define COLS_PER_SPLIT 1024
#define TILES_PER_SPLIT 64   // 1024 / 16
#define COL_SPLITS 8
#define ROWS_PER_BLOCK 128   // 8 waves * 16 rows
#define BPAD 264             // 256 + 8 f16 pad -> 528B row stride, banks 4n
#define LOG2E 1.44269504f

union AFrag { v16h v; v8h h[2]; };

// async-DMA builtin takes (v4i addrspace(1)*, v4i addrspace(3)*, imm, imm)
#define AS1P(p) ((__attribute__((address_space(1))) v4i*)(p))
#define AS3P(p) ((__attribute__((address_space(3))) v4i*)(p))

// ---------------------------------------------------------------------------
// Kernel 1: L2-normalize rows of [z_i ; z_j], emit f16 zn [8192][256].
// ---------------------------------------------------------------------------
__global__ __launch_bounds__(256)
void ntxent_normalize(const float* __restrict__ z_i,
                      const float* __restrict__ z_j,
                      f16* __restrict__ zn)
{
    int wave = threadIdx.x >> 5, lane = threadIdx.x & 31;
    int row  = blockIdx.x * 8 + wave;
    const float* src = (row < B_ROWS) ? (z_i + (size_t)row * DIM)
                                      : (z_j + (size_t)(row - B_ROWS) * DIM);
    v4f x0 = *(const v4f*)(src + lane * 8);
    v4f x1 = *(const v4f*)(src + lane * 8 + 4);
    float ss = 0.f;
#pragma unroll
    for (int k = 0; k < 4; ++k) ss += x0[k] * x0[k] + x1[k] * x1[k];
#pragma unroll
    for (int off = 16; off; off >>= 1) ss += __shfl_xor(ss, off, 32);
    float inv = 1.0f / fmaxf(sqrtf(ss), 1e-8f);
    v8h o;
#pragma unroll
    for (int k = 0; k < 4; ++k) {
        o[k]     = (f16)(x0[k] * inv);
        o[k + 4] = (f16)(x1[k] * inv);
    }
    *(v8h*)(zn + (size_t)row * DIM + lane * 8) = o;
}

// ---------------------------------------------------------------------------
// Kernel 2: fused sim-GEMM + masked exp row-accumulation.
// grid = (64 row-blocks, 8 col-splits), block = 256 threads (8 waves).
// B tiles (16 cols x 256 K, 8 KB) are DMA'd global->LDS with
// GLOBAL_LOAD_ASYNC_TO_LDS_B128, double-buffered, shared by all 8 waves.
// Each wave keeps its 16-row A fragments (16x256 f16) in VGPRs for the
// whole sweep; 8x v_wmma_f32_16x16x32_f16 per tile; branch-free epilogue.
// Partial row sums written per col-split (no atomics -> deterministic).
// ---------------------------------------------------------------------------
__global__ __launch_bounds__(256)
void ntxent_sim(const f16* __restrict__ zn,
                const int* __restrict__ labels,
                float* __restrict__ g_part)
{
    __shared__ __attribute__((aligned(16))) f16 s_b[2][16 * BPAD];
    __shared__ int s_lab[COLS_PER_SPLIT];

    int tid  = threadIdx.x;
    int wave = tid >> 5, lane = tid & 31;
    int half = lane >> 4, l16 = lane & 15;
    int row_base = blockIdx.x * ROWS_PER_BLOCK + wave * 16;
    int col0     = blockIdx.y * COLS_PER_SPLIT;

#pragma unroll
    for (int i = 0; i < COLS_PER_SPLIT / 256; ++i)
        s_lab[tid + i * 256] = labels[(col0 + tid + i * 256) & (B_ROWS - 1)];

    // Preload A fragments (CDNA5 16-bit A 16x32 lane swizzle).
    AFrag a[KSTEPS];
    {
        const f16* ap = zn + (size_t)(row_base + l16) * DIM + half * 8;
#pragma unroll
        for (int ks = 0; ks < KSTEPS; ++ks) {
            a[ks].h[0] = *(const v8h*)(ap + ks * 32);
            a[ks].h[1] = *(const v8h*)(ap + ks * 32 + 16);
        }
    }
    int labr[8];
#pragma unroll
    for (int r = 0; r < 8; ++r)
        labr[r] = labels[(row_base + 8 * half + r) & (B_ROWS - 1)];

    // Async-copy mapping: thread -> (col 0..15, 32B chunk); 2x B128 per thread.
    int cp_col = tid >> 4;                 // 0..15
    int cp_off = (tid & 15) * 16;          // f16 elements, 16 per thread

    float acc[8] = {0.f, 0.f, 0.f, 0.f, 0.f, 0.f, 0.f, 0.f};

    // Prologue: prefetch tile 0 into buffer 0.
    {
        const f16* g = zn + (size_t)(col0 + cp_col) * DIM + cp_off;
        f16* l = &s_b[0][cp_col * BPAD + cp_off];
        __builtin_amdgcn_global_load_async_to_lds_b128(AS1P(g),     AS3P(l),     0, 0);
        __builtin_amdgcn_global_load_async_to_lds_b128(AS1P(g + 8), AS3P(l + 8), 0, 0);
    }

    for (int t = 0; t < TILES_PER_SPLIT; ++t) {
        int cur = t & 1;
        if (t + 1 < TILES_PER_SPLIT) {
            const f16* g = zn + (size_t)(col0 + (t + 1) * 16 + cp_col) * DIM + cp_off;
            f16* l = &s_b[cur ^ 1][cp_col * BPAD + cp_off];
            __builtin_amdgcn_global_load_async_to_lds_b128(AS1P(g),     AS3P(l),     0, 0);
            __builtin_amdgcn_global_load_async_to_lds_b128(AS1P(g + 8), AS3P(l + 8), 0, 0);
            asm volatile("s_wait_asynccnt 0x2" ::: "memory");  // tile t landed
        } else {
            asm volatile("s_wait_asynccnt 0x0" ::: "memory");
        }
        __syncthreads();   // all waves' copies for tile t visible

        // B 32x16 fragment from LDS: lanes 0-15 K=k0..k0+7,k0+16..k0+23 (col=lane),
        // lanes 16-31 K=k0+8..k0+15,k0+24..k0+31.
        const f16* bp = &s_b[cur][l16 * BPAD] + half * 8;
        v8f c = {};
#pragma unroll
        for (int ks = 0; ks < KSTEPS; ++ks) {
            AFrag b;
            b.h[0] = *(const v8h*)(bp + ks * 32);
            b.h[1] = *(const v8h*)(bp + ks * 32 + 16);
            c = __builtin_amdgcn_wmma_f32_16x16x32_f16(
                    false, a[ks].v, false, b.v, (short)0, c, false, false);
        }

        // Branch-free epilogue: s in [-2,2] -> fixed lse shift of 2.
        int lab_c = s_lab[t * 16 + l16];
#pragma unroll
        for (int r = 0; r < 8; ++r) {
            float s = c[r] * 2.0f;                       // / TEMPERATURE
            float e = __builtin_amdgcn_exp2f((s - 2.0f) * LOG2E);
            acc[r] += (labr[r] != lab_c) ? e : 0.0f;
        }
        __syncthreads();   // done reading s_b[cur] before tile t+2 overwrites
    }

    // Reduce each row's partial across the 16 lanes of its half-wave.
#pragma unroll
    for (int r = 0; r < 8; ++r) {
        float v = acc[r];
#pragma unroll
        for (int off = 1; off < 16; off <<= 1) v += __shfl_xor(v, off, 32);
        if (l16 == 0)
            g_part[(size_t)blockIdx.y * N_ROWS + row_base + 8 * half + r] = v;
    }
}

// ---------------------------------------------------------------------------
// Kernel 3: per-row loss. One wave per row: recompute pos as a dot product,
// sum the 8 col-split partials in fixed order (deterministic).
// ---------------------------------------------------------------------------
__global__ __launch_bounds__(256)
void ntxent_rowloss(const f16* __restrict__ zn,
                    const float* __restrict__ g_part,
                    float* __restrict__ g_loss)
{
    int wave = threadIdx.x >> 5, lane = threadIdx.x & 31;
    int i = blockIdx.x * 8 + wave;
    const v8h va = *(const v8h*)(zn + (size_t)i * DIM + lane * 8);
    const v8h vb = *(const v8h*)(zn + (size_t)(i ^ B_ROWS) * DIM + lane * 8);
    float d = 0.f;
#pragma unroll
    for (int k = 0; k < 8; ++k) d += (float)va[k] * (float)vb[k];
#pragma unroll
    for (int off = 16; off; off >>= 1) d += __shfl_xor(d, off, 32);
    float pos = 2.0f * d;                                // / TEMPERATURE
    float accv = 0.f;
#pragma unroll
    for (int s = 0; s < COL_SPLITS; ++s)
        accv += g_part[(size_t)s * N_ROWS + i];
    float lse = 2.0f + logf(__builtin_amdgcn_exp2f((pos - 2.0f) * LOG2E) + accv);
    if (lane == 0) g_loss[i] = lse - pos;
}

// ---------------------------------------------------------------------------
// Kernel 4: deterministic single-block reduction of 8192 row losses.
// ---------------------------------------------------------------------------
__global__ __launch_bounds__(1024)
void ntxent_reduce(const float* __restrict__ g_loss, float* __restrict__ out)
{
    __shared__ float red[32];
    int tid = threadIdx.x;
    float sum = 0.f;
#pragma unroll
    for (int i = 0; i < N_ROWS / 1024; ++i)
        sum += g_loss[tid + i * 1024];
#pragma unroll
    for (int off = 16; off; off >>= 1) sum += __shfl_xor(sum, off, 32);
    if ((tid & 31) == 0) red[tid >> 5] = sum;
    __syncthreads();
    if (tid < 32) {
        float v = red[tid];
#pragma unroll
        for (int off = 16; off; off >>= 1) v += __shfl_xor(v, off, 32);
        if (tid == 0) out[0] = v / (float)N_ROWS;
    }
}

// ---------------------------------------------------------------------------
extern "C" void kernel_launch(void* const* d_in, const int* in_sizes, int n_in,
                              void* d_out, int out_size, void* d_ws, size_t ws_size,
                              hipStream_t stream) {
    const float* z_i    = (const float*)d_in[0];
    const float* z_j    = (const float*)d_in[1];
    const int*   labels = (const int*)d_in[2];
    float* out = (float*)d_out;

    // workspace layout (all fully rewritten every launch; no memset needed)
    f16*   zn     = (f16*)d_ws;                                         // 4 MB
    float* g_part = (float*)((char*)d_ws + (size_t)N_ROWS * DIM * sizeof(f16));
    float* g_loss = g_part + (size_t)COL_SPLITS * N_ROWS;               // +256 KB

    ntxent_normalize<<<N_ROWS / 8, 256, 0, stream>>>(z_i, z_j, zn);

    dim3 grid(N_ROWS / ROWS_PER_BLOCK, COL_SPLITS);                     // 64 x 8
    ntxent_sim<<<grid, 256, 0, stream>>>(zn, labels, g_part);

    ntxent_rowloss<<<N_ROWS / 8, 256, 0, stream>>>(zn, g_part, g_loss);
    ntxent_reduce<<<1, 1024, 0, stream>>>(g_loss, out);
}
